// TKFFTConv_3470333575751
// MI455X (gfx1250) — matrix operand order repaired and checked by reading, not compiled, not run on hardware
//
#include <hip/hip_runtime.h>

// ---------------------------------------------------------------------------
// Monarch FFT conv for MI455X (gfx1250, wave32, WMMA bf16 16x16x32).
// Memory-bound problem (~205MB @ 23.3TB/s ~ 9us): one fused pass over u,
// 48 v_wmma_f32_16x16x32_bf16 per 32x32 tile, all repacks via per-wave LDS.
// v2: 8 tiles per wave (amortize const-table init), packed b128 transposed
// repack stores, global_prefetch of the next tile.
// ---------------------------------------------------------------------------

typedef __bf16 v16bf __attribute__((ext_vector_type(16)));
typedef __bf16 v8bf  __attribute__((ext_vector_type(8)));
typedef float  v8f   __attribute__((ext_vector_type(8)));

#define TWO_PI 6.283185307179586f
#define H_CH   768
#define BATCH  64
#define NTILES (BATCH * H_CH)   // 49152
#define TPW    8                // tiles per wave

// D = A(16x32 bf16) x B(32x16 bf16) + C(16x16 f32)
__device__ __forceinline__ v8f wmma_bf16(v16bf a, v16bf b, v8f c) {
  return __builtin_amdgcn_wmma_f32_16x16x32_bf16(false, a, false, b, (short)0, c,
                                                 false, false);
}

// A-fragment (16x32, M-rows [16q,16q+15]) from a row-major 32x32 bf16 matrix.
// Lane layout (ISA 7.12.2): lanes 0-15 & 16-31 share M=lane%16; half h=lane/16
// gets K = {8h..8h+7, 16+8h..16+8h+7}  -> two ds_load_b128.
__device__ __forceinline__ v16bf loadA(const __bf16* mat, int q, int lane) {
  int m = lane & 15, h = lane >> 4;
  const __bf16* p = mat + ((16 * q + m) << 5) + (h << 3);
  v16bf r;
  *(v8bf*)&r     = *(const v8bf*)p;         // K = 8h .. 8h+7
  ((v8bf*)&r)[1] = *(const v8bf*)(p + 16);  // K = 16+8h .. 16+8h+7
  return r;
}

// Store a C/D accumulator quadrant (qm,qn) into a row-major 32x32 bf16 matrix.
// C layout: elem v lives at M = 16*qm + 8*(lane/16) + v, N = 16*qn + lane%16.
// Column-order addresses (stride 64B) -> 8 scalar b16 stores.
__device__ __forceinline__ void storeC(__bf16* mat, v8f a, int qm, int qn,
                                       int lane) {
  int n = (qn << 4) + (lane & 15);
  int h = lane >> 4;
#pragma unroll
  for (int v = 0; v < 8; ++v) {
    int m = (qm << 4) + (h << 3) + v;
    mat[(m << 5) + n] = (__bf16)a[v];
  }
}

// Transposed store: the lane's 8 values are contiguous along the fast axis
// (16B, 16B-aligned) -> convert to v8bf, single ds_store_b128.
__device__ __forceinline__ void storeCT(__bf16* mat, v8f a, int qm, int qn,
                                        int lane) {
  int n = (qn << 4) + (lane & 15);
  int h = lane >> 4;
  int m0 = (qm << 4) + (h << 3);
  v8bf q;
#pragma unroll
  for (int v = 0; v < 8; ++v) q[v] = (__bf16)a[v];
  *(v8bf*)(mat + (n << 5) + m0) = q;
}

// ---------------------------------------------------------------------------
// Prologue 1: Z = (F32 @ Kmat) * exp(-2pi i * i*j/1024), fp32. One thread per
// (h,i,j).  Kmat[a,j] = k[h, a*32+j].
// ---------------------------------------------------------------------------
__global__ void __launch_bounds__(256)
filter_fft_p1(const float* __restrict__ k, float* __restrict__ Zr,
              float* __restrict__ Zi) {
  int t = blockIdx.x * 256 + threadIdx.x;  // H*1024 threads
  int h = t >> 10, m = t & 1023;
  int i = m >> 5, j = m & 31;
  const float* kp = k + (h << 10) + j;
  float zr = 0.f, zi = 0.f;
#pragma unroll
  for (int a = 0; a < 32; ++a) {
    float kv = kp[a << 5];
    float ang = -TWO_PI * (float)((i * a) & 31) * (1.0f / 32.0f);
    float s, c;
    __sincosf(ang, &s, &c);
    zr += c * kv;
    zi += s * kv;
  }
  float ang = -TWO_PI * (float)((i * j) & 1023) * (1.0f / 1024.0f);
  float s, c;
  __sincosf(ang, &s, &c);
  Zr[t] = zr * c - zi * s;
  Zi[t] = zr * s + zi * c;
}

// ---------------------------------------------------------------------------
// Prologue 2: X = Z @ F32 ; bf16-round; scatter into per-lane C-layout
// fragment order so the main kernel reads k_fT with contiguous 32B loads.
// kTC layout: [h][qm][qn][lane][ r0..r7 | i0..i7 ]  (bf16).
// ---------------------------------------------------------------------------
__global__ void __launch_bounds__(256)
filter_fft_p2(const float* __restrict__ Zr, const float* __restrict__ Zi,
              __bf16* __restrict__ kTC) {
  int t = blockIdx.x * 256 + threadIdx.x;
  int h = t >> 10, m = t & 1023;
  int i = m >> 5, j = m & 31;
  const float* zr = Zr + (h << 10) + (i << 5);
  const float* zi = Zi + (h << 10) + (i << 5);
  float xr = 0.f, xi = 0.f;
#pragma unroll
  for (int a = 0; a < 32; ++a) {
    float ang = -TWO_PI * (float)((a * j) & 31) * (1.0f / 32.0f);
    float s, c;
    __sincosf(ang, &s, &c);
    float r = zr[a], q = zi[a];
    xr += r * c - q * s;
    xi += r * s + q * c;
  }
  int qm = i >> 4, qn = j >> 4;
  int ln = (j & 15) + (((i & 15) >> 3) << 4);
  int v = i & 7;
  size_t base = ((size_t)h * 4 + (size_t)(qm * 2 + qn)) * 512 + (size_t)ln * 16;
  kTC[base + v] = (__bf16)xr;
  kTC[base + 8 + v] = (__bf16)xi;
}

// ---------------------------------------------------------------------------
// Main fused kernel: 8 waves/block, TPW tiles per wave.
// Pipeline (A operand is ALWAYS the intermediate, B is ALWAYS a constant;
// F/Finv symmetric so left-multiplies become transposed right-multiplies,
// transposes folded into the LDS repack addressing):
//   Y1  = U^T @ F           (transposed state), * tw,   store^T  -> T1'
//   T2  = T1' @ F,          * k_fT,                     store    -> T3
//   T4  = T3  @ Finv,       * twinv,                    store^T  -> T4^T
//   O^T = Re(T4^T @ Finv),                              store^T  -> out
// ---------------------------------------------------------------------------
__global__ void __launch_bounds__(256)
fftconv_main(const __bf16* __restrict__ u, const __bf16* __restrict__ kTC,
             __bf16* __restrict__ out) {
  // B fragments of the constant DFT matrices, per lane:
  //  0,1: F_re (qn=0,1)   2,3: F_im   4,5: -F_im
  //  6,7: G_re (G=Finv)   8,9: G_im  10,11: -G_im
  __shared__ __align__(32) __bf16 sBfrag[12][32][16];
  // Per-lane twiddle factors in C layout: [mat(tw,twinv)][qm][qn][lane][r8,i8]
  __shared__ float sTw[2][2][2][32][16];
  // Per-wave repack buffers (row-major 32x32 bf16).
  __shared__ __align__(16) __bf16 sRe[8][1024];
  __shared__ __align__(16) __bf16 sIm[8][1024];

  // ---- build constants (bf16-rounded, matching the reference) ----
  for (int idx = threadIdx.x; idx < 12 * 32; idx += 256) {
    int frag = idx >> 5, ln = idx & 31;
    int mat = frag / 6, sub = frag % 6;
    int comp = sub >> 1, qn = sub & 1;
    int kh = (ln >> 4) << 4;              // K base: kh/2 = 8*(lane/16)
    int nn = (ln & 15) + (qn << 4);
#pragma unroll
    for (int e = 0; e < 16; ++e) {
      int K = (kh >> 1) + e + ((e >> 3) << 3);
      float ang = TWO_PI * (float)((K * nn) & 31) * (1.0f / 32.0f);
      float s, c;
      __sincosf(ang, &s, &c);
      float val;
      if (comp == 0)      val = c;                 // Re
      else if (comp == 1) val = mat ? s : -s;      // F_im = -sin, G_im = +sin
      else                val = mat ? -s : s;      // negated imag
      sBfrag[frag][ln][e] = (__bf16)val;
    }
  }
  for (int idx = threadIdx.x; idx < 2 * 2 * 2 * 32; idx += 256) {
    int mat = idx >> 7;
    int qm = (idx >> 6) & 1, qn = (idx >> 5) & 1, ln = idx & 31;
    int nn = (qn << 4) + (ln & 15);
    int mb = (qm << 4) + ((ln >> 4) << 3);
#pragma unroll
    for (int v = 0; v < 8; ++v) {
      int m = mb + v;
      float ang = TWO_PI * (float)((m * nn) & 1023) * (1.0f / 1024.0f);
      float s, c;
      __sincosf(ang, &s, &c);
      float rr, ii;
      if (mat == 0) { rr = c * (1.0f / 1024.0f); ii = -s * (1.0f / 1024.0f); }
      else          { rr = c;                    ii = s; }
      sTw[mat][qm][qn][ln][v]     = (float)(__bf16)rr;  // bf16-round, use f32
      sTw[mat][qm][qn][ln][8 + v] = (float)(__bf16)ii;
    }
  }
  __syncthreads();

  int lane = threadIdx.x & 31;
  int wave = threadIdx.x >> 5;
  int tile0 = (blockIdx.x * 8 + wave) * TPW;
  if (tile0 >= NTILES) return;

  __bf16* re = sRe[wave];
  __bf16* im = sIm[wave];

  const __bf16* up = u + (size_t)tile0 * 1024;
  __bf16* op = out + (size_t)tile0 * 1024;
  int h = tile0 % H_CH;

  for (int it = 0; it < TPW; ++it) {
    // prefetch next tile's u region (64B per lane covers the 2KB tile)
    if (it + 1 < TPW)
      __builtin_prefetch((const void*)(up + 1024 + lane * 32), 0, 1);

    // filter spectrum fragments for this channel (L2-resident, 32B/lane/quad)
    const __bf16* kp = kTC + (size_t)h * 2048 + (size_t)lane * 16;
    v16bf kq[2][2];
#pragma unroll
    for (int qm = 0; qm < 2; ++qm)
#pragma unroll
      for (int qn = 0; qn < 2; ++qn)
        kq[qm][qn] = *(const v16bf*)(kp + ((qm * 2 + qn) << 9));

    // ---- load U (coalesced b128) and store transposed: LDS holds U^T ----
#pragma unroll
    for (int i = 0; i < 4; ++i) {
      int e0 = i * 256 + lane * 8;
      v8bf val = *(const v8bf*)(up + e0);
      int r = e0 >> 5, c0 = e0 & 31;
#pragma unroll
      for (int e = 0; e < 8; ++e) re[((c0 + e) << 5) + r] = val[e];
    }

    // ---- Stage 1: Y1 = U^T @ F, * tw (transposed), store^T -> T1' ----
    {
      v16bf ua0 = loadA(re, 0, lane);
      v16bf ua1 = loadA(re, 1, lane);
#pragma unroll
      for (int qm = 0; qm < 2; ++qm) {
        v16bf ua = qm ? ua1 : ua0;
#pragma unroll
        for (int qn = 0; qn < 2; ++qn) {
          v16bf fBr = *(const v16bf*)sBfrag[0 + qn][lane];
          v16bf fBi = *(const v16bf*)sBfrag[2 + qn][lane];
          v8f z = {};
          v8f yr = wmma_bf16(ua, fBr, z);
          v8f yi = wmma_bf16(ua, fBi, z);
          const float* tp = sTw[0][qm][qn][lane];
          v8f xr, xi;
#pragma unroll
          for (int v = 0; v < 8; ++v) {
            float tr = tp[v], ti = tp[8 + v];
            xr[v] = yr[v] * tr - yi[v] * ti;
            xi[v] = yr[v] * ti + yi[v] * tr;
          }
          storeCT(re, xr, qm, qn, lane);
          storeCT(im, xi, qm, qn, lane);
        }
      }
    }

    // ---- Stage 2: T2 = T1' @ F, * k_fT, store -> T3 ----
    {
      v16bf ar0 = loadA(re, 0, lane), ar1 = loadA(re, 1, lane);
      v16bf ai0 = loadA(im, 0, lane), ai1 = loadA(im, 1, lane);
#pragma unroll
      for (int qm = 0; qm < 2; ++qm) {
        v16bf ar = qm ? ar1 : ar0;
        v16bf ai = qm ? ai1 : ai0;
#pragma unroll
        for (int qn = 0; qn < 2; ++qn) {
          v16bf fBr  = *(const v16bf*)sBfrag[0 + qn][lane];
          v16bf fBi  = *(const v16bf*)sBfrag[2 + qn][lane];
          v16bf fBni = *(const v16bf*)sBfrag[4 + qn][lane];
          v8f z = {};
          v8f trr = wmma_bf16(ai, fBni, wmma_bf16(ar, fBr, z));
          v8f tii = wmma_bf16(ai, fBr,  wmma_bf16(ar, fBi, z));
          v8f xr, xi;
#pragma unroll
          for (int v = 0; v < 8; ++v) {
            float kr = (float)kq[qm][qn][v];
            float ki = (float)kq[qm][qn][8 + v];
            xr[v] = trr[v] * kr - tii[v] * ki;
            xi[v] = trr[v] * ki + tii[v] * kr;
          }
          storeC(re, xr, qm, qn, lane);
          storeC(im, xi, qm, qn, lane);
        }
      }
    }

    // ---- Stage 3: T4 = T3 @ Finv, * twinv, store^T -> T4^T ----
    {
      v16bf ar0 = loadA(re, 0, lane), ar1 = loadA(re, 1, lane);
      v16bf ai0 = loadA(im, 0, lane), ai1 = loadA(im, 1, lane);
#pragma unroll
      for (int qm = 0; qm < 2; ++qm) {
        v16bf ar = qm ? ar1 : ar0;
        v16bf ai = qm ? ai1 : ai0;
#pragma unroll
        for (int qn = 0; qn < 2; ++qn) {
          v16bf gBr  = *(const v16bf*)sBfrag[6 + qn][lane];
          v16bf gBi  = *(const v16bf*)sBfrag[8 + qn][lane];
          v16bf gBni = *(const v16bf*)sBfrag[10 + qn][lane];
          v8f z = {};
          v8f trr = wmma_bf16(ai, gBni, wmma_bf16(ar, gBr, z));
          v8f tii = wmma_bf16(ai, gBr,  wmma_bf16(ar, gBi, z));
          const float* tp = sTw[1][qm][qn][lane];
          v8f xr, xi;
#pragma unroll
          for (int v = 0; v < 8; ++v) {
            float tr = tp[v], ti = tp[8 + v];
            xr[v] = trr[v] * tr - tii[v] * ti;
            xi[v] = trr[v] * ti + tii[v] * tr;
          }
          storeCT(re, xr, qm, qn, lane);
          storeCT(im, xi, qm, qn, lane);
        }
      }
    }

    // ---- Stage 4: O^T = Re(T4^T @ Finv), store^T -> out (row-major) ----
    {
      v16bf ar0 = loadA(re, 0, lane), ar1 = loadA(re, 1, lane);
      v16bf ai0 = loadA(im, 0, lane), ai1 = loadA(im, 1, lane);
#pragma unroll
      for (int qm = 0; qm < 2; ++qm) {
        v16bf ar = qm ? ar1 : ar0;
        v16bf ai = qm ? ai1 : ai0;
#pragma unroll
        for (int qn = 0; qn < 2; ++qn) {
          v16bf gBr  = *(const v16bf*)sBfrag[6 + qn][lane];
          v16bf gBni = *(const v16bf*)sBfrag[10 + qn][lane];
          v8f z = {};
          v8f o = wmma_bf16(ai, gBni, wmma_bf16(ar, gBr, z));
          storeCT(re, o, qm, qn, lane);
        }
      }
    }

    // ---- coalesced store of the tile ----
#pragma unroll
    for (int i = 0; i < 4; ++i) {
      int e0 = i * 256 + lane * 8;
      *(v8bf*)(op + e0) = *(const v8bf*)(re + e0);
    }

    up += 1024;
    op += 1024;
    if (++h == H_CH) h = 0;
  }
}

// ---------------------------------------------------------------------------
extern "C" void kernel_launch(void* const* d_in, const int* in_sizes, int n_in,
                              void* d_out, int out_size, void* d_ws,
                              size_t ws_size, hipStream_t stream) {
  const __bf16* u = (const __bf16*)d_in[0];   // (64,768,1024) bf16
  const float* k  = (const float*)d_in[1];    // (768,1024) f32
  __bf16* out     = (__bf16*)d_out;           // (64,768,1024) bf16

  char* ws = (char*)d_ws;
  float*  Zr  = (float*)ws;                                   // 3 MB
  float*  Zi  = (float*)(ws + (size_t)H_CH * 1024 * 4);       // 3 MB
  __bf16* kTC = (__bf16*)(ws + (size_t)2 * H_CH * 1024 * 4);  // 3 MB

  int pgrid = (H_CH * 1024) / 256;  // 3072
  filter_fft_p1<<<pgrid, 256, 0, stream>>>(k, Zr, Zi);
  filter_fft_p2<<<pgrid, 256, 0, stream>>>(Zr, Zi, kTC);

  int mgrid = NTILES / (8 * TPW);  // 768 blocks, 8 waves, TPW tiles per wave
  fftconv_main<<<mgrid, 256, 0, stream>>>(u, kTC, out);
}
